// NeuralSplineCoupling_8014408974444
// MI455X (gfx1250) — matrix operand order repaired and checked by reading, not compiled, not run on hardware
//
#include <hip/hip_runtime.h>
#include <hip/hip_bf16.h>
#include <math.h>

// ---------------------------------------------------------------------------
// NeuralSplineCoupling forward for MI455X (gfx1250, wave32, WMMA + TDM).
//  prep kernel : fold BN into W1/b1, transpose + f16-ify W1/W2/W3 into d_ws.
//  main kernel : TDM (tensor_load_to_lds) stages all f16 weights into LDS,
//                async b32 loads stage bias tables; per-wave 16-row tiles run
//                3 GEMMs on v_wmma_f32_16x16x32_f16; RQS spline epilogue in
//                fp32 VALU + TRANS (v_exp/v_log/v_rcp) ops.
// ---------------------------------------------------------------------------

typedef _Float16 half_t;
typedef __attribute__((ext_vector_type(16))) _Float16 v16h;
typedef __attribute__((ext_vector_type(8)))  float    v8f;
typedef __attribute__((ext_vector_type(4)))  float    f32x4;

#define HID     128
#define P_PAD   384          // 376 spline cols padded to 24 WMMA tiles
#define BN_EPS  1e-5f
#define WAVES   4            // waves per block
#define TILES   4            // 16-row tiles per wave
#define ROWS_PER_BLOCK (WAVES * 16 * TILES)   // 256

#if defined(__gfx1250__)
#define HAS_TDM __has_builtin(__builtin_amdgcn_tensor_load_to_lds)
#else
#define HAS_TDM 0
#endif

union FragH { v16h v; f32x4 q[2]; };

__device__ __forceinline__ void wave_ldsbar() {
  // wave-private LDS RAW/WAR fence: order DS ops + stop compiler reordering
  __asm volatile("s_wait_dscnt 0x0" ::: "memory");
}

__device__ __forceinline__ float fast_rcp(float v) {
#if defined(__AMDGCN__)
  return __builtin_amdgcn_rcpf(v);   // v_rcp_f32 (TRANS pipe, 1 instr)
#else
  return 1.0f / v;
#endif
}
__device__ __forceinline__ float swishf(float v) {
  return v * fast_rcp(1.0f + __expf(-v));
}
__device__ __forceinline__ float softplusf(float v) {
  return (v > 20.0f) ? v : __logf(1.0f + __expf(v));
}

#if HAS_TDM
typedef __attribute__((ext_vector_type(4))) unsigned int u32x4;
typedef __attribute__((ext_vector_type(8))) int          i32x8;
typedef __attribute__((ext_vector_type(4))) int          i32x4;

// Issue one TDM DMA: contiguous `bytes` from global -> LDS, as a 1-D tile of
// 8-byte elements (D# layout per CDNA5 ISA ch.8; type=2, count=1).
__device__ __forceinline__ void tdm_load_1d(unsigned lds_off, const void* gsrc,
                                            unsigned bytes) {
  unsigned long long ga = (unsigned long long)(uintptr_t)gsrc;
  unsigned n8 = bytes >> 3;                       // elements of 8 bytes
  u32x4 g0;
  g0[0] = 1u;                                     // count=1 (valid user D#)
  g0[1] = lds_off;                                // lds_addr  [63:32]
  g0[2] = (unsigned)ga;                           // global_addr[31:0]
  g0[3] = (unsigned)((ga >> 32) & 0x01FFFFFFu)    // global_addr[56:32]
          | (2u << 30);                           // type=2 ("image")
  i32x8 g1;
  g1[0] = (int)(3u << 16);                        // data_size=3 (8B), mask=0
  g1[1] = (int)((n8 & 0xFFFFu) << 16);            // tensor_dim0[15:0]
  g1[2] = (int)((n8 >> 16) & 0xFFFFu)             // tensor_dim0[31:16]
          | (1 << 16);                            // tensor_dim1 = 1
  g1[3] = (int)((n8 & 0xFFFFu) << 16);            // tile_dim0 = n8
  g1[4] = 0;                                      // tile_dim1/2 unused
  g1[5] = (int)n8;                                // tensor_dim0_stride
  g1[6] = 0;
  g1[7] = 0;
  i32x4 z4 = {0, 0, 0, 0};
#if __clang_major__ >= 23
  i32x8 z8 = {0, 0, 0, 0, 0, 0, 0, 0};
  __builtin_amdgcn_tensor_load_to_lds(g0, g1, z4, z4, z8, 0);
#else
  __builtin_amdgcn_tensor_load_to_lds(g0, g1, z4, z4, 0);
#endif
}
#endif // HAS_TDM

// ---------------------------------------------------------------------------
// Prep: BN fold + transpose + f16 conversion into workspace.
//   w1t [128][32]  f16 : W1'[k][n] = W1[k][n]*a[k], K padded 16->32 with 0
//   w2t [128][128] f16 : W2^T (K contiguous per output col)
//   w3t [384][128] f16 : W3^T, cols 376..383 zero
//   b1f [128] f32 : b1 + bvec @ W1 ;  b3f [384] f32 : b3 zero-padded
// ---------------------------------------------------------------------------
__global__ void nsc_prep(const float* __restrict__ bn_scale,
                         const float* __restrict__ bn_bias,
                         const float* __restrict__ bn_mean,
                         const float* __restrict__ bn_var,
                         const float* __restrict__ W1, const float* __restrict__ b1,
                         const float* __restrict__ W2,
                         const float* __restrict__ W3, const float* __restrict__ b3,
                         half_t* __restrict__ w1t, half_t* __restrict__ w2t,
                         half_t* __restrict__ w3t,
                         float* __restrict__ b1f, float* __restrict__ b3f) {
  const int tid = blockIdx.x * blockDim.x + threadIdx.x;
  const int nth = gridDim.x * blockDim.x;

  for (int i = tid; i < 128 * 32; i += nth) {
    int nn = i >> 5, k = i & 31;
    float v = 0.0f;
    if (k < 16) {
      float a = bn_scale[k] * rsqrtf(bn_var[k] + BN_EPS);
      v = W1[k * HID + nn] * a;
    }
    w1t[i] = (half_t)v;
  }
  for (int i = tid; i < HID; i += nth) {
    float s = b1[i];
    #pragma unroll
    for (int k = 0; k < 16; ++k) {
      float a  = bn_scale[k] * rsqrtf(bn_var[k] + BN_EPS);
      float bv = bn_bias[k] - bn_mean[k] * a;
      s += bv * W1[k * HID + i];
    }
    b1f[i] = s;
  }
  for (int i = tid; i < HID * HID; i += nth) {
    int nn = i >> 7, k = i & 127;
    w2t[i] = (half_t)W2[k * HID + nn];
  }
  for (int i = tid; i < P_PAD * HID; i += nth) {
    int nn = i >> 7, k = i & 127;
    w3t[i] = (half_t)((nn < 376) ? W3[k * 376 + nn] : 0.0f);
  }
  for (int i = tid; i < P_PAD; i += nth) b3f[i] = (i < 376) ? b3[i] : 0.0f;
}

// ---------------------------------------------------------------------------
// Main kernel.
// ---------------------------------------------------------------------------
__launch_bounds__(WAVES * 32, 1)
__global__ void nsc_main(const float* __restrict__ x, const float* __restrict__ cc,
                         const half_t* __restrict__ w1t_g,
                         const half_t* __restrict__ w2t_g,
                         const half_t* __restrict__ w3t_g,
                         const float* __restrict__ b1f, const float* __restrict__ b2,
                         const float* __restrict__ b3f,
                         float* __restrict__ out, int n) {
  __shared__ half_t sW1[128 * 32];          //   8 KB
  __shared__ half_t sW2[128 * 128];         //  32 KB
  __shared__ half_t sW3[P_PAD * 128];       //  96 KB
  __shared__ float  sB1[128];
  __shared__ float  sB2[128];
  __shared__ float  sB3[P_PAD];
  __shared__ half_t sA[WAVES * 16 * 128];   //  16 KB per-wave activation staging
  __shared__ float  sP[WAVES * 16 * P_PAD]; //  96 KB per-wave spline params
  __shared__ float  sLd[WAVES * 16];

  const int tid    = threadIdx.x;
  const int wid    = tid >> 5;
  const int lane   = tid & 31;
  const int half16 = lane >> 4;
  const int l16    = lane & 15;

  // ---- stage bias tables via async load-to-LDS (ASYNCcnt path) -----------
#if defined(__gfx1250__)
  {
    int i = tid;   // blockDim.x == 128, one element per thread for b1/b2
    {
      unsigned lo1 = (unsigned)(uintptr_t)&sB1[i];
      unsigned long long ga1 = (unsigned long long)(uintptr_t)(b1f + i);
      __asm volatile("global_load_async_to_lds_b32 %0, %1, off"
                     :: "v"(lo1), "v"(ga1) : "memory");
      unsigned lo2 = (unsigned)(uintptr_t)&sB2[i];
      unsigned long long ga2 = (unsigned long long)(uintptr_t)(b2 + i);
      __asm volatile("global_load_async_to_lds_b32 %0, %1, off"
                     :: "v"(lo2), "v"(ga2) : "memory");
    }
    for (int j = tid; j < P_PAD; j += (int)blockDim.x) {
      unsigned lo3 = (unsigned)(uintptr_t)&sB3[j];
      unsigned long long ga3 = (unsigned long long)(uintptr_t)(b3f + j);
      __asm volatile("global_load_async_to_lds_b32 %0, %1, off"
                     :: "v"(lo3), "v"(ga3) : "memory");
    }
    __asm volatile("s_wait_asynccnt 0x0" ::: "memory");
  }
#else
  for (int i = tid; i < 128;   i += blockDim.x) { sB1[i] = b1f[i]; sB2[i] = b2[i]; }
  for (int i = tid; i < P_PAD; i += blockDim.x) sB3[i] = b3f[i];
#endif

  // ---- stage f16 weights via Tensor Data Mover (TENSORcnt path) ----------
#if HAS_TDM
  if (wid == 0) {
    tdm_load_1d((unsigned)(uintptr_t)&sW1[0], w1t_g, 128 * 32 * 2);
    tdm_load_1d((unsigned)(uintptr_t)&sW2[0], w2t_g, 128 * 128 * 2);
    tdm_load_1d((unsigned)(uintptr_t)&sW3[0], w3t_g, P_PAD * 128 * 2);
    __builtin_amdgcn_s_wait_tensorcnt(0);
  }
#else
  for (int i = tid; i < 128 * 32;    i += blockDim.x) sW1[i] = w1t_g[i];
  for (int i = tid; i < 128 * 128;   i += blockDim.x) sW2[i] = w2t_g[i];
  for (int i = tid; i < P_PAD * 128; i += blockDim.x) sW3[i] = w3t_g[i];
#endif

  // ---- warm L2 with this block's x / c rows (global_prefetch_b8) ---------
  {
    size_t row0 = (size_t)blockIdx.x * ROWS_PER_BLOCK;
    if (row0 + ROWS_PER_BLOCK <= (size_t)n) {
      const char* px = (const char*)(x + row0 * 16);
      __builtin_prefetch(px + tid * 128, 0, 1);          // 16 KB of x
      const char* pc = (const char*)(cc + row0 * 8);
      if (tid < 64) __builtin_prefetch(pc + tid * 128, 0, 1);  // 8 KB of c
    }
  }
  __syncthreads();

  half_t* sAw  = sA  + wid * 16 * 128;
  float*  sPw  = sP  + wid * 16 * P_PAD;
  float*  sLdw = sLd + wid * 16;

  float* yout  = out;
  float* ldout = out + (size_t)n * 16;

  const int waveRow0 = blockIdx.x * ROWS_PER_BLOCK + wid * (16 * TILES);

  for (int it = 0; it < TILES; ++it) {
    const int tileBase = waveRow0 + it * 16;
    if (tileBase >= n) break;                 // wave-uniform
    const int rowg = tileBase + l16;
    const int rowc = (rowg < n) ? rowg : (n - 1);   // clamp: keep EXEC all-1s

    // ---- A0 fragment: h = [xcond | c] (K=16, padded to 32) --------------
    // A layout (16-bit, 16x32): lane half 0 holds K=0..7 (+16..23),
    //                           lane half 1 holds K=8..15 (+24..31)
    FragH A0;
    {
      const float* src = half16 ? (cc + (size_t)rowc * 8)
                                : (x  + (size_t)rowc * 16 + 8);
      f32x4 u0 = *(const f32x4*)src;
      f32x4 u1 = *(const f32x4*)(src + 4);
      #pragma unroll
      for (int j = 0; j < 4; ++j) A0.v[j]     = (half_t)u0[j];
      #pragma unroll
      for (int j = 0; j < 4; ++j) A0.v[4 + j] = (half_t)u1[j];
      #pragma unroll
      for (int j = 8; j < 16; ++j) A0.v[j] = (half_t)0.0f;
    }

    // ---- GEMM1: h1 = swish(A0 * W1 + b1) -> sAw (f16 [16][128]) ---------
    #pragma unroll
    for (int t = 0; t < 8; ++t) {
      FragH B;
      const half_t* bp = sW1 + (t * 16 + l16) * 32 + half16 * 8;
      B.q[0] = *(const f32x4*)bp;
      B.q[1] = *(const f32x4*)(bp + 16);
      v8f acc = {};
      acc = __builtin_amdgcn_wmma_f32_16x16x32_f16(false, A0.v, false, B.v,
                                                   (short)0, acc, false, false);
      float bias = sB1[t * 16 + l16];
      #pragma unroll
      for (int r = 0; r < 8; ++r) {
        float v = swishf(acc[r] + bias);
        sAw[(r + 8 * half16) * 128 + t * 16 + l16] = (half_t)v;
      }
    }
    wave_ldsbar();

    // ---- GEMM2: h2 = swish(h1 * W2 + b2), K=128 -> 4 chained WMMA -------
    v8f acc2[8];
    #pragma unroll
    for (int t = 0; t < 8; ++t) {
      v8f acc = {};
      #pragma unroll
      for (int kb = 0; kb < 4; ++kb) {
        FragH A, B;
        const half_t* ap = sAw + l16 * 128 + kb * 32 + half16 * 8;
        A.q[0] = *(const f32x4*)ap;
        A.q[1] = *(const f32x4*)(ap + 16);
        const half_t* bp = sW2 + (t * 16 + l16) * 128 + kb * 32 + half16 * 8;
        B.q[0] = *(const f32x4*)bp;
        B.q[1] = *(const f32x4*)(bp + 16);
        acc = __builtin_amdgcn_wmma_f32_16x16x32_f16(false, A.v, false, B.v,
                                                     (short)0, acc, false, false);
      }
      acc2[t] = acc;
    }
    wave_ldsbar();     // all reads of sAw done before overwrite
    #pragma unroll
    for (int t = 0; t < 8; ++t) {
      float bias = sB2[t * 16 + l16];
      #pragma unroll
      for (int r = 0; r < 8; ++r) {
        float v = swishf(acc2[t][r] + bias);
        sAw[(r + 8 * half16) * 128 + t * 16 + l16] = (half_t)v;
      }
    }
    wave_ldsbar();

    // ---- GEMM3: p = h2 * W3 + b3 -> sPw (f32 [16][384]); W3 in LDS ------
    for (int t = 0; t < 24; ++t) {
      v8f acc = {};
      #pragma unroll
      for (int kb = 0; kb < 4; ++kb) {
        FragH A, B;
        const half_t* ap = sAw + l16 * 128 + kb * 32 + half16 * 8;
        A.q[0] = *(const f32x4*)ap;
        A.q[1] = *(const f32x4*)(ap + 16);
        const half_t* bp = sW3 + (t * 16 + l16) * 128 + kb * 32 + half16 * 8;
        B.q[0] = *(const f32x4*)bp;
        B.q[1] = *(const f32x4*)(bp + 16);
        acc = __builtin_amdgcn_wmma_f32_16x16x32_f16(false, A.v, false, B.v,
                                                     (short)0, acc, false, false);
      }
      float bias = sB3[t * 16 + l16];
      #pragma unroll
      for (int r = 0; r < 8; ++r)
        sPw[(r + 8 * half16) * P_PAD + t * 16 + l16] = acc[r] + bias;
    }

    if (lane < 16) sLdw[lane] = 0.0f;
    wave_ldsbar();

    // ---- RQS spline epilogue: 128 (row,d) tasks per tile, 4 per lane ----
    for (int tt = 0; tt < 4; ++tt) {
      const int task = lane + 32 * tt;
      const int rl = task >> 3;                  // local row 0..15
      const int d  = task & 7;                   // transform dim 0..7
      const int rg = tileBase + rl;
      const int rc = (rg < n) ? rg : (n - 1);
      const float* p = sPw + rl * P_PAD + d * 47;

      float xt  = x[(size_t)rc * 16 + d];
      bool  oob = (xt < 0.0f) || (xt > 1.0f);
      float xcv = fminf(fmaxf(xt, 0.0f), 1.0f);

      float m1 = -1e30f, m2 = -1e30f;
      #pragma unroll 4
      for (int j = 0; j < 16; ++j) { m1 = fmaxf(m1, p[j]); m2 = fmaxf(m2, p[16 + j]); }
      float s1 = 0.0f, s2 = 0.0f;
      #pragma unroll 4
      for (int j = 0; j < 16; ++j) { s1 += __expf(p[j] - m1); s2 += __expf(p[16 + j] - m2); }
      float i1 = fast_rcp(s1), i2 = fast_rcp(s2);

      // streaming cumsum + bin search over internal knots
      float cx = 0.0f, cy = 0.0f, xk0 = 0.0f, yk0 = 0.0f;
      float wk = __expf(p[0]  - m1) * i1;
      float hk = __expf(p[16] - m2) * i2;
      int   k  = 0;
      for (int i = 0; i < 15; ++i) {
        float dxi = __expf(p[i]      - m1) * i1;
        float dyi = __expf(p[16 + i] - m2) * i2;
        cx += dxi; cy += dyi;
        if (cx <= xcv) {
          k = i + 1; xk0 = cx; yk0 = cy;
          wk = __expf(p[i + 1]      - m1) * i1;
          hk = __expf(p[16 + i + 1] - m2) * i2;
        }
      }
      float d0 = (k == 0)  ? 1.0f : softplusf(p[32 + k - 1]);
      float d1 = (k == 15) ? 1.0f : softplusf(p[32 + k]);

      float iwk = fast_rcp(wk);
      float xi  = (xcv - xk0) * iwk;
      float s   = hk * iwk;
      float z   = xi * (1.0f - xi);
      float den = s + (d1 + d0 - 2.0f * s) * z;
      float iden = fast_rcp(den);
      float yv  = yk0 + hk * (s * xi * xi + d0 * z) * iden;
      float om  = 1.0f - xi;
      float dv  = s * s * (d1 * xi * xi + 2.0f * s * z + d0 * om * om) * iden * iden;
      yv = oob ? xt : yv;
      float ld = oob ? 0.0f : __logf(dv);

      if (rg < n) {
        yout[(size_t)rg * 16 + d]     = yv;                          // yt
        yout[(size_t)rg * 16 + 8 + d] = x[(size_t)rg * 16 + 8 + d];  // xcond copy
        atomicAdd(&sLdw[rl], ld);                                    // ds_add_f32
      }
    }
    wave_ldsbar();
    if (lane < 16) {
      int rg = tileBase + lane;
      if (rg < n) ldout[rg] = sLdw[lane];
    }
    wave_ldsbar();   // sPw / sLdw safe to overwrite next tile
  }
}

// ---------------------------------------------------------------------------
extern "C" void kernel_launch(void* const* d_in, const int* in_sizes, int n_in,
                              void* d_out, int out_size, void* d_ws, size_t ws_size,
                              hipStream_t stream) {
  const float* x        = (const float*)d_in[0];
  const float* c        = (const float*)d_in[1];
  const float* bn_scale = (const float*)d_in[2];
  const float* bn_bias  = (const float*)d_in[3];
  const float* bn_mean  = (const float*)d_in[4];
  const float* bn_var   = (const float*)d_in[5];
  const float* W1       = (const float*)d_in[6];
  const float* b1       = (const float*)d_in[7];
  const float* W2       = (const float*)d_in[8];
  const float* b2       = (const float*)d_in[9];
  const float* W3       = (const float*)d_in[10];
  const float* b3       = (const float*)d_in[11];
  float* out = (float*)d_out;
  const int n = in_sizes[0] / 16;

  // workspace layout (f16 weights + f32 biases), ~142 KB
  half_t* w1t = (half_t*)d_ws;            // 128*32
  half_t* w2t = w1t + 128 * 32;           // 128*128
  half_t* w3t = w2t + 128 * 128;          // 384*128
  float*  b1f = (float*)(w3t + P_PAD * HID);
  float*  b3f = b1f + HID;

  nsc_prep<<<64, 256, 0, stream>>>(bn_scale, bn_bias, bn_mean, bn_var,
                                   W1, b1, W2, W3, b3,
                                   w1t, w2t, w3t, b1f, b3f);

  const int grid = (n + ROWS_PER_BLOCK - 1) / ROWS_PER_BLOCK;
  nsc_main<<<grid, WAVES * 32, 0, stream>>>(x, c, w1t, w2t, w3t,
                                            b1f, b2, b3f, out, n);
  (void)n_in; (void)out_size; (void)ws_size;
}
